// update_e2_25950192403283
// MI455X (gfx1250) — compile-verified, matrix operand order checked by reading, not run on hardware
//
#include <hip/hip_runtime.h>
#include <hip/hip_bf16.h>
#include <math.h>

typedef __attribute__((ext_vector_type(16))) __bf16 v16bf;
typedef __attribute__((ext_vector_type(8)))  __bf16 v8bf;
typedef __attribute__((ext_vector_type(4)))  __bf16 v4bf;
typedef __attribute__((ext_vector_type(8)))  float  v8f;

#define HIDDEN   128
#define NOUT     256            // 2*HIDDEN
#define KIN      306            // 2*HIDDEN + 50
#define KPAD     320            // K padded to multiple of 32 (10 WMMA steps)
#define KSTRIDE  328            // LDS row stride (bank-conflict-free for ds_load_b128)
#define YSTR     260            // Ys row stride (conflict-free)
#define NGAUSS   50
#define TILE_M   64             // edges per workgroup
#define BN_EPS   1e-5f
#define PI_OVER_CUT 0.31415926535897932f   // pi / 10

// ---------------------------------------------------------------------------
// Kernel 0: convert + swizzle W (256 x 306 f32, [N][K] row-major) into WMMA
// B-fragment order:  block fb = kk*16 + nt  (1 KB each, 512 bf16);
// within a block, lane L owns 16 bf16 at element offset L*16:
//   elements 0..7  = W[n][kb .. kb+7]
//   elements 8..15 = W[n][kb+16 .. kb+23]
// with n = nt*16 + (L&15), kb = kk*32 + (L>>4)*8.  K >= 306 zero-padded.
// Also zeroes the stat accumulators (first 512 floats of ws).
// ---------------------------------------------------------------------------
__global__ void prep_kernel(const float* __restrict__ W,
                            __bf16* __restrict__ Wswz,
                            float* __restrict__ stats) {
    int idx = blockIdx.x * blockDim.x + threadIdx.x;
    if (idx < NOUT * KPAD) {
        int fb = idx >> 9;            // fragment block
        int w  = idx & 511;
        int L  = w >> 4;              // lane 0..31
        int c  = w & 15;              // element within lane
        int kk = fb >> 4;
        int nt = fb & 15;
        int n  = nt * 16 + (L & 15);
        int k  = kk * 32 + (L >> 4) * 8 + ((c < 8) ? c : c + 8);
        float val = (k < KIN) ? W[n * KIN + k] : 0.0f;
        Wswz[idx] = (__bf16)val;
    }
    if (idx < 2 * NOUT) stats[idx] = 0.0f;
}

// ---------------------------------------------------------------------------
// Kernel 2: finalize BN params. scale = gamma*rsqrt(var+eps),
// shift = (bias - mean)*scale + beta   (bias folded; GEMM acc excludes bias)
// ---------------------------------------------------------------------------
__global__ void finalize_kernel(const float* __restrict__ stats,
                                const float* __restrict__ bias,
                                const float* __restrict__ gamma,
                                const float* __restrict__ beta,
                                float* __restrict__ scale,
                                float* __restrict__ shift,
                                int E) {
    int n = threadIdx.x;
    if (n < NOUT) {
        float invE = 1.0f / (float)E;
        float mean = stats[n] * invE;
        float var  = stats[NOUT + n] * invE - mean * mean;
        float sc   = gamma[n] * rsqrtf(var + BN_EPS);
        scale[n] = sc;
        shift[n] = (bias[n] - mean) * sc + beta[n];
    }
}

// ---------------------------------------------------------------------------
// Kernels 1 & 3: tile GEMM via v_wmma_f32_16x16x32_bf16.
// phase 0: accumulate per-feature sum / sumsq (x includes bias).
// phase 1: y = acc*scale + shift, gated activation + cosine cutoff -> out.
// 256 threads = 8 waves. Wave w owns M-tile (w>>1), N-half (w&1): 16x128 slab.
// ---------------------------------------------------------------------------
__global__ void edge_gemm_kernel(const float* __restrict__ v,
                                 const float* __restrict__ dist,
                                 const float* __restrict__ dist_emb,
                                 const long long* __restrict__ edge_index,
                                 const __bf16* __restrict__ Wswz,
                                 const float* __restrict__ bias,
                                 const float* __restrict__ scale,
                                 const float* __restrict__ shift,
                                 float* __restrict__ gstats,
                                 float* __restrict__ out,
                                 int E, int phase) {
    __shared__ __bf16 As[TILE_M * KSTRIDE];    // ~41 KB gathered inputs (bf16)
    __shared__ float  Ys[TILE_M * YSTR];       // ~65 KB pre-activation tile
    __shared__ float  shsum[NOUT];
    __shared__ float  shsq[NOUT];
    __shared__ float  Cenv[TILE_M];
    __shared__ int    sII[TILE_M];
    __shared__ int    sJJ[TILE_M];

    const int t  = threadIdx.x;
    const int e0 = blockIdx.x * TILE_M;

    if (t < NOUT) { shsum[t] = 0.0f; shsq[t] = 0.0f; }

    // ---- cache edge indices + envelope ----
    if (t < TILE_M) {
        const int e = e0 + t;
        const bool valid = (e < E);
        sJJ[t] = valid ? (int)edge_index[e] : 0;                  // edge_index[0][e]
        sII[t] = valid ? (int)edge_index[(long long)E + e] : 0;   // edge_index[1][e]
        float d = valid ? dist[e] : 0.0f;
        Cenv[t] = 0.5f * (__cosf(d * PI_OVER_CUT) + 1.0f);
    }
    __syncthreads();

    // ---- stage A tile (vectorized float4 gathers, bf16 into LDS) ----
    {
        const float4* v4 = (const float4*)v;   // 32 float4 per node row
        #pragma unroll
        for (int it = 0; it < 8; ++it) {       // v[i] section -> k 0..127
            int lin = it * 256 + t;            // 0..2047
            int r   = lin >> 5;
            int c4  = lin & 31;
            float4 val = v4[(long long)sII[r] * 32 + c4];
            v4bf pk = { (__bf16)val.x, (__bf16)val.y, (__bf16)val.z, (__bf16)val.w };
            *reinterpret_cast<v4bf*>(&As[r * KSTRIDE + c4 * 4]) = pk;
        }
        #pragma unroll
        for (int it = 0; it < 8; ++it) {       // v[j] section -> k 128..255
            int lin = it * 256 + t;
            int r   = lin >> 5;
            int c4  = lin & 31;
            float4 val = v4[(long long)sJJ[r] * 32 + c4];
            v4bf pk = { (__bf16)val.x, (__bf16)val.y, (__bf16)val.z, (__bf16)val.w };
            *reinterpret_cast<v4bf*>(&As[r * KSTRIDE + HIDDEN + c4 * 4]) = pk;
        }
        for (int idx = t; idx < TILE_M * NGAUSS; idx += 256) {   // k 256..305
            int r = idx / NGAUSS;
            int c = idx - r * NGAUSS;
            float val = (e0 + r < E) ? dist_emb[(long long)(e0 + r) * NGAUSS + c] : 0.0f;
            As[r * KSTRIDE + 2 * HIDDEN + c] = (__bf16)val;
        }
        for (int idx = t; idx < TILE_M * (KSTRIDE - KIN); idx += 256) {  // zero pad
            int r = idx / (KSTRIDE - KIN);
            int c = idx - r * (KSTRIDE - KIN);
            As[r * KSTRIDE + KIN + c] = (__bf16)0.0f;
        }
    }
    __syncthreads();

    // ---- WMMA tile GEMM ----
    const int wave   = t >> 5;
    const int lane   = t & 31;
    const int mtile  = wave >> 1;        // 0..3
    const int nhalf  = wave & 1;         // 0..1
    const int laneN  = lane & 15;
    const int laneHi = lane >> 4;        // selects K chunk {0-7,16-23} vs {8-15,24-31}
    const int rowA   = mtile * 16 + laneN;
    const int koff   = laneHi * 8;

    // coalesced fragment base: lane L reads [base + L*16 .. +15] bf16
    const __bf16* wbase = Wswz + (nhalf * 8) * 512 + lane * 16;

    v8f acc[8];
    #pragma unroll
    for (int nt = 0; nt < 8; ++nt) acc[nt] = (v8f)0.0f;

    #pragma unroll
    for (int kk = 0; kk < KPAD / 32; ++kk) {
        const int kb = kk * 32 + koff;
        v8bf alo = *reinterpret_cast<const v8bf*>(&As[rowA * KSTRIDE + kb]);
        v8bf ahi = *reinterpret_cast<const v8bf*>(&As[rowA * KSTRIDE + kb + 16]);
        v16bf afrag = __builtin_shufflevector(alo, ahi,
            0,1,2,3,4,5,6,7,8,9,10,11,12,13,14,15);
        const __bf16* wk = wbase + kk * 16 * 512;
        #pragma unroll
        for (int nt = 0; nt < 8; ++nt) {
            const __bf16* wp = wk + nt * 512;
            v8bf blo = *reinterpret_cast<const v8bf*>(wp);
            v8bf bhi = *reinterpret_cast<const v8bf*>(wp + 8);
            v16bf bfrag = __builtin_shufflevector(blo, bhi,
                0,1,2,3,4,5,6,7,8,9,10,11,12,13,14,15);
            acc[nt] = __builtin_amdgcn_wmma_f32_16x16x32_bf16(
                false, afrag, false, bfrag, (short)0, acc[nt], false, false);
        }
    }

    // C/D layout: lane holds N = laneN (+tile offsets); element vv is M = laneHi*8 + vv.
    const int rowBase = mtile * 16 + laneHi * 8;

    if (phase == 0) {
        // ---- batch statistics ----
        #pragma unroll
        for (int nt = 0; nt < 8; ++nt) {
            const int n = nhalf * 128 + nt * 16 + laneN;
            const float bn = bias[n];
            float s = 0.0f, q = 0.0f;
            #pragma unroll
            for (int vv = 0; vv < 8; ++vv) {
                if (e0 + rowBase + vv < E) {
                    float xv = acc[nt][vv] + bn;
                    s += xv;
                    q += xv * xv;
                }
            }
            atomicAdd(&shsum[n], s);   // ds_add_f32
            atomicAdd(&shsq[n],  q);
        }
        __syncthreads();
        if (t < NOUT) {
            atomicAdd(&gstats[t], shsum[t]);          // global_atomic_add_f32
            atomicAdd(&gstats[NOUT + t], shsq[t]);
        }
    } else {
        // ---- BN affine -> LDS, fused gated activation + cutoff ----
        #pragma unroll
        for (int nt = 0; nt < 8; ++nt) {
            const int n = nhalf * 128 + nt * 16 + laneN;
            const float sc = scale[n];
            const float sh = shift[n];
            #pragma unroll
            for (int vv = 0; vv < 8; ++vv) {
                Ys[(rowBase + vv) * YSTR + n] = acc[nt][vv] * sc + sh;
            }
        }
        __syncthreads();
        for (int idx = t; idx < TILE_M * HIDDEN; idx += 256) {
            const int m = idx >> 7;
            const int h = idx & (HIDDEN - 1);
            const int e = e0 + m;
            if (e < E) {
                float yc = Ys[m * YSTR + h];
                float yf = Ys[m * YSTR + HIDDEN + h];
                float sp = (yc > 15.0f) ? yc : log1pf(__expf(yc));   // softplus
                float sg = 1.0f / (1.0f + __expf(-yf));              // sigmoid
                out[(long long)e * HIDDEN + h] = sp * sg * Cenv[m];
            }
        }
    }
}

// ---------------------------------------------------------------------------
extern "C" void kernel_launch(void* const* d_in, const int* in_sizes, int n_in,
                              void* d_out, int out_size, void* d_ws, size_t ws_size,
                              hipStream_t stream) {
    const float*     v        = (const float*)d_in[0];
    const float*     dist     = (const float*)d_in[1];
    const float*     dist_emb = (const float*)d_in[2];
    const long long* eidx     = (const long long*)d_in[3];
    const float*     W        = (const float*)d_in[4];
    const float*     b        = (const float*)d_in[5];
    const float*     gamma    = (const float*)d_in[6];
    const float*     beta     = (const float*)d_in[7];
    float*           out      = (float*)d_out;

    const int E = in_sizes[1];               // n_edges

    // workspace layout
    float*  ws_f   = (float*)d_ws;
    float*  gstats = ws_f;                   // sum[256] + sumsq[256]
    float*  scale  = ws_f + 512;
    float*  shift  = ws_f + 768;
    __bf16* Wswz   = (__bf16*)((char*)d_ws + 4096);   // 256*320 bf16 = 160 KB, swizzled

    const int nblk = (E + TILE_M - 1) / TILE_M;

    // 0) convert + swizzle weights, zero stats
    prep_kernel<<<(NOUT * KPAD + 255) / 256, 256, 0, stream>>>(W, Wswz, gstats);
    // 1) GEMM pass for batch statistics
    edge_gemm_kernel<<<nblk, 256, 0, stream>>>(v, dist, dist_emb, eidx, Wswz, b,
                                               scale, shift, gstats, out, E, 0);
    // 2) finalize BN affine params
    finalize_kernel<<<1, 256, 0, stream>>>(gstats, b, gamma, beta, scale, shift, E);
    // 3) GEMM pass + BN + gated activation + cutoff -> output
    edge_gemm_kernel<<<nblk, 256, 0, stream>>>(v, dist, dist_emb, eidx, Wswz, b,
                                               scale, shift, gstats, out, E, 1);
}
